// SceneGraphModel_72980084293700
// MI455X (gfx1250) — matrix operand order repaired and checked by reading, not compiled, not run on hardware
//
#include <hip/hip_runtime.h>

typedef _Float16 v16h __attribute__((ext_vector_type(16)));
typedef _Float16 v8h  __attribute__((ext_vector_type(8)));
typedef float    v8f  __attribute__((ext_vector_type(8)));

#define WAVES_PER_BLOCK 4
#define TPB (WAVES_PER_BLOCK * 32)
#define TILES 2  // 16-edge tiles per wave iteration (share B fragments)

// Packed weight-fragment offsets in halves (each fragment = 32 lanes * 16 halves = 512)
enum : int {
  WO1 = 0,      // 16->128 (+bias row k=16) : 1 kchunk * 8 nchunks = 8 frags
  WO2 = 4096,   // 128->64 : 4 * 4 = 16 frags
  WO3 = 12288,  // 64->32  : 2 * 2 = 4 frags
  WO4 = 14336,  // 32->16  : 1
  WO5 = 14848,  // 16->8  (+bias row k=16) : 1
  WO6 = 15360,  // 8->4   (+bias row k=8)  : 1
  WO7 = 15872,  // 4->50  (+bias row k=4)  : 1 * 4 = 4 frags
  W_TOTAL = 17920
};
enum : int {
  BIAS_BYTES_OFF = W_TOTAL * 2,              // 35840 ; only b2/b3/b4 live here
  BIAS_BYTES     = 512,                      // 112 floats used, padded
  ACT_BYTES_OFF  = BIAS_BYTES_OFF + BIAS_BYTES,  // 36352
  // per-tile activation buffers (halves): A0[16][32] A1[16][128] A2[16][64]
  // A3[16][32] A4[16][32] A5[16][32] A6[16][32]
  ACT_TILE       = 5632,                     // halves per 16-edge tile slot
  ACT_PER_WAVE   = TILES * ACT_TILE          // 11264 halves
};
#define SMEM_BYTES (ACT_BYTES_OFF + WAVES_PER_BLOCK * ACT_PER_WAVE * 2)  // 126464 B

// layer offsets inside one tile slot (halves)
enum : int { AO0 = 0, AO1 = 512, AO2 = 2560, AO3 = 3584, AO4 = 4096, AO5 = 4608, AO6 = 5120 };

__device__ __forceinline__ void wait_ds() {
  asm volatile("s_wait_dscnt 0" ::: "memory");
}

// B fragment (32x16 f16, K x N): lane<16 -> column N=lane, K=0..7,16..23;
// lane>=16 -> column N=lane-16, K=8..15,24..31. Pre-packed contiguously per lane.
__device__ __forceinline__ v16h load_b_frag(const _Float16* p) {
  v8h lo = *(const v8h*)p;
  v8h hi = *(const v8h*)(p + 8);
  v16h r;
#pragma unroll
  for (int i = 0; i < 8; ++i) { r[i] = lo[i]; r[8 + i] = hi[i]; }
  return r;
}

// A fragment (16x32 f16, M x K): lane row m=lane&15; lane half h=lane>>4:
//   elements 0..7  = K = k0 + 8h + 0..7
//   elements 8..15 = K = k0 + 16 + 8h + 0..7
__device__ __forceinline__ v16h load_a_frag(const _Float16* actIn, int stride,
                                            int k0, int m0, int h) {
  const _Float16* p = actIn + m0 * stride + k0 + 8 * h;
  v8h lo = *(const v8h*)p;
  v8h hi = *(const v8h*)(p + 16);
  v16h r;
#pragma unroll
  for (int i = 0; i < 8; ++i) { r[i] = lo[i]; r[8 + i] = hi[i]; }
  return r;
}

// One dense layer over TILES 16-edge tiles. Accumulators start from the
// WMMA inline-0 C operand; bias comes either from a folded ones-column in
// the padded K range (EPI_BIAS=false) or a fused add in the epilogue.
// ReLU is applied in f16 after conversion (commutes with monotonic cvt);
// cvt outputs are canonical, so the f16 maxnum needs no canonicalize and
// both cvt and max pack (v_cvt_pk_f16_f32 / v_pk_max_num_f16).
template <int KCH, int NCH, int IN_STRIDE, int OUT_STRIDE, bool RELU, int NACT, bool EPI_BIAS>
__device__ __forceinline__ void mlp_layer(const _Float16* actIn, _Float16* actOut,
                                          const _Float16* wfrag, const float* biasp) {
  const int lane = threadIdx.x & 31;
  const int m0 = lane & 15;
  const int h = lane >> 4;
  wait_ds();  // prior layer's cross-lane LDS stores must land
  v16h afr[TILES][KCH];
#pragma unroll
  for (int t = 0; t < TILES; ++t)
#pragma unroll
    for (int kc = 0; kc < KCH; ++kc)
      afr[t][kc] = load_a_frag(actIn + t * ACT_TILE, IN_STRIDE, kc * 32, m0, h);

  const v8f zc = {};
#pragma unroll
  for (int nc = 0; nc < NCH; ++nc) {
    const int n = nc * 16 + m0;
    float bv = 0.0f;
    if (EPI_BIAS) bv = biasp[n];  // NACT == 16*NCH for all epilogue-bias layers
    v8f acc[TILES];
#pragma unroll
    for (int kc = 0; kc < KCH; ++kc) {
      v16h bfr = load_b_frag(wfrag + (kc * NCH + nc) * 512 + lane * 16);
#pragma unroll
      for (int t = 0; t < TILES; ++t) {
        acc[t] = __builtin_amdgcn_wmma_f32_16x16x32_f16(
            false, afr[t][kc], false, bfr, (short)0, kc == 0 ? zc : acc[t],
            false, false);
      }
    }
#pragma unroll
    for (int t = 0; t < TILES; ++t) {
      if (NACT == 16 * NCH || n < NACT) {
        _Float16* ao = actOut + t * ACT_TILE;
        v8h hv;
#pragma unroll
        for (int r = 0; r < 8; ++r) {
          float v = acc[t][r];
          if (EPI_BIAS) v += bv;
          hv[r] = (_Float16)v;  // adjacent pairs fuse to v_cvt_pk_f16_f32
        }
        if (RELU) {
#pragma unroll
          for (int r = 0; r < 8; ++r)
            hv[r] = __builtin_fmaxf16(hv[r], (_Float16)0.0f);  // packs to v_pk_max
        }
#pragma unroll
        for (int r = 0; r < 8; ++r)
          ao[(r + 8 * h) * OUT_STRIDE + n] = hv[r];  // row m = r + 8h
      }
    }
  }
}

// Convert W[k][n] (row-major f32) into per-lane B-fragment f16 layout.
// Optionally appends bias[n] as row k==Kact (must fall in the zero-padded K range).
__device__ __forceinline__ void pack_weights(_Float16* dst, const float* W,
                                             const float* bias_row, int Kact,
                                             int Nact, int kch, int nch) {
  const int total = kch * nch * 512;
  for (int t = threadIdx.x; t < total; t += TPB) {
    const int f = t >> 9;
    const int w = t & 511;
    const int lane = w >> 4;
    const int i = w & 15;
    const int kc = f / nch;
    const int nc = f - kc * nch;
    const int n = nc * 16 + (lane & 15);
    const int k = kc * 32 + 8 * (lane >> 4) + (i & 7) + ((i >> 3) << 4);
    float v = 0.0f;
    if (n < Nact) {
      if (k < Kact) v = W[k * Nact + n];
      else if (k == Kact && bias_row) v = bias_row[n];
    }
    dst[t] = (_Float16)v;
  }
}

__global__ __launch_bounds__(TPB) void sg_edge_mlp_kernel(
    const float* __restrict__ x, const int* __restrict__ eidx,
    const float* __restrict__ W1, const float* __restrict__ b1,
    const float* __restrict__ W2, const float* __restrict__ b2,
    const float* __restrict__ W3, const float* __restrict__ b3,
    const float* __restrict__ W4, const float* __restrict__ b4,
    const float* __restrict__ W5, const float* __restrict__ b5,
    const float* __restrict__ W6, const float* __restrict__ b6,
    const float* __restrict__ W7, const float* __restrict__ b7,
    float* __restrict__ out, int E, int nTiles) {
  extern __shared__ char smem[];
  _Float16* wp = (_Float16*)smem;
  float* bias = (float*)(smem + BIAS_BYTES_OFF);
  _Float16* actbase = (_Float16*)(smem + ACT_BYTES_OFF);

  const int lane = threadIdx.x & 31;
  const int wave = threadIdx.x >> 5;
  const int m0 = lane & 15;
  const int h = lane >> 4;

  // ---- one-time per block: pack weights (+folded bias rows) into LDS ----
  pack_weights(wp + WO1, W1, b1, 16, 128, 1, 8);
  pack_weights(wp + WO2, W2, nullptr, 128, 64, 4, 4);
  pack_weights(wp + WO3, W3, nullptr, 64, 32, 2, 2);
  pack_weights(wp + WO4, W4, nullptr, 32, 16, 1, 1);
  pack_weights(wp + WO5, W5, b5, 16, 8, 1, 1);
  pack_weights(wp + WO6, W6, b6, 8, 4, 1, 1);
  pack_weights(wp + WO7, W7, b7, 4, 50, 1, 4);
  for (int i = threadIdx.x; i < 112; i += TPB) {  // epilogue biases: b2,b3,b4
    float v;
    if (i < 64)       v = b2[i];
    else if (i < 96)  v = b3[i - 64];
    else              v = b4[i - 96];
    bias[i] = v;
  }

  // ---- wave-private activation slots; zero once, then plant ones-columns ----
  _Float16* act = actbase + wave * ACT_PER_WAVE;
  {
    v8h z = {};
    v8h* av = (v8h*)act;
    for (int i = lane; i < ACT_PER_WAVE / 8; i += 32) av[i] = z;
  }
  wait_ds();
  if (lane < 16) {
#pragma unroll
    for (int t = 0; t < TILES; ++t) {
      _Float16* a = act + t * ACT_TILE;
      a[AO0 + lane * 32 + 16] = (_Float16)1.0f;  // layer1 bias input
      a[AO4 + lane * 32 + 16] = (_Float16)1.0f;  // layer5 bias input
      a[AO5 + lane * 32 + 8]  = (_Float16)1.0f;  // layer6 bias input
      a[AO6 + lane * 32 + 4]  = (_Float16)1.0f;  // layer7 bias input
    }
  }
  __syncthreads();

  const int wslot = blockIdx.x * WAVES_PER_BLOCK + wave;
  const int wstride = gridDim.x * WAVES_PER_BLOCK;

  for (int tile = wslot; tile < nTiles; tile += wstride) {
    const int tileBase = tile * (16 * TILES);

    // ---- gather: lane handles edge (base + m0), endpoint h, per tile slot ----
#pragma unroll
    for (int t = 0; t < TILES; ++t) {
      int e = tileBase + t * 16 + m0;
      int ec = e < E ? e : (E - 1);
      long long idx = (long long)eidx[(size_t)h * (size_t)E + ec];
      const float4* xr = (const float4*)(x + (size_t)idx * 8);
      float4 f0 = xr[0];
      float4 f1 = xr[1];
      v8h hv;
      hv[0] = (_Float16)f0.x; hv[1] = (_Float16)f0.y;
      hv[2] = (_Float16)f0.z; hv[3] = (_Float16)f0.w;
      hv[4] = (_Float16)f1.x; hv[5] = (_Float16)f1.y;
      hv[6] = (_Float16)f1.z; hv[7] = (_Float16)f1.w;
      // concat [src, dst]: src -> cols 0..7 (h==0), dst -> cols 8..15 (h==1)
      *(v8h*)(act + t * ACT_TILE + AO0 + m0 * 32 + 8 * h) = hv;
    }

    // ---- MLP layers via WMMA (bias folded or fused; C starts at inline 0) ----
    mlp_layer<1, 8, 32, 128, true, 128, false>(act + AO0, act + AO1, wp + WO1, nullptr);
    mlp_layer<4, 4, 128, 64, true, 64,  true >(act + AO1, act + AO2, wp + WO2, bias + 0);
    mlp_layer<2, 2, 64, 32,  true, 32,  true >(act + AO2, act + AO3, wp + WO3, bias + 64);
    mlp_layer<1, 1, 32, 32,  false, 16, true >(act + AO3, act + AO4, wp + WO4, bias + 96);
    mlp_layer<1, 1, 32, 32,  true, 8,   false>(act + AO4, act + AO5, wp + WO5, nullptr);
    mlp_layer<1, 1, 32, 32,  true, 4,   false>(act + AO5, act + AO6, wp + WO6, nullptr);

    // ---- layer 7: [16x4+1s pad32] @ [(4+bias)x50 pad 32x64] -> global f32 ----
    {
      wait_ds();
      v16h afr[TILES];
#pragma unroll
      for (int t = 0; t < TILES; ++t)
        afr[t] = load_a_frag(act + t * ACT_TILE + AO6, 32, 0, m0, h);
      const v8f zc = {};
#pragma unroll
      for (int nc = 0; nc < 4; ++nc) {
        const int n = nc * 16 + m0;
        v16h bfr = load_b_frag(wp + WO7 + nc * 512 + lane * 16);
#pragma unroll
        for (int t = 0; t < TILES; ++t) {
          v8f acc = __builtin_amdgcn_wmma_f32_16x16x32_f16(false, afr[t], false,
                                                           bfr, (short)0, zc,
                                                           false, false);
          if (n < 50) {
#pragma unroll
            for (int r = 0; r < 8; ++r) {
              const int edge = tileBase + t * 16 + r + 8 * h;
              if (edge < E) out[(size_t)edge * 50 + n] = acc[r];
            }
          }
        }
      }
    }
  }
}

__global__ void copy_y_kernel(const int* __restrict__ y, int* __restrict__ dst, int n) {
  int i = blockIdx.x * blockDim.x + threadIdx.x;
  if (i < n) dst[i] = y[i];
}

extern "C" void kernel_launch(void* const* d_in, const int* in_sizes, int n_in,
                              void* d_out, int out_size, void* d_ws, size_t ws_size,
                              hipStream_t stream) {
  const float* x = (const float*)d_in[0];
  const int* y = (const int*)d_in[1];
  const int* eidx = (const int*)d_in[2];
  const float* W1 = (const float*)d_in[3];
  const float* b1 = (const float*)d_in[4];
  const float* W2 = (const float*)d_in[5];
  const float* b2 = (const float*)d_in[6];
  const float* W3 = (const float*)d_in[7];
  const float* b3 = (const float*)d_in[8];
  const float* W4 = (const float*)d_in[9];
  const float* b4 = (const float*)d_in[10];
  const float* W5 = (const float*)d_in[11];
  const float* b5 = (const float*)d_in[12];
  const float* W6 = (const float*)d_in[13];
  const float* b6 = (const float*)d_in[14];
  const float* W7 = (const float*)d_in[15];
  const float* b7 = (const float*)d_in[16];
  float* out = (float*)d_out;

  const int E = in_sizes[1];  // y has one label per edge
  const int nTiles = (E + (16 * TILES - 1)) / (16 * TILES);
  int blocks = (nTiles + WAVES_PER_BLOCK - 1) / WAVES_PER_BLOCK;
  if (blocks > 3072) blocks = 3072;
  if (blocks < 1) blocks = 1;

  sg_edge_mlp_kernel<<<blocks, TPB, SMEM_BYTES, stream>>>(
      x, eidx, W1, b1, W2, b2, W3, b3, W4, b4, W5, b5, W6, b6, W7, b7,
      out, E, nTiles);

  // Pass-through y into the tail of d_out (after the E*50 logits).
  const long long tail = (long long)out_size - (long long)E * 50;
  if (tail > 0) {
    int ncopy = (tail >= 2LL * E) ? 2 * E : (int)((tail < (long long)E) ? tail : E);
    int* dsty = (int*)d_out + (size_t)E * 50;
    copy_y_kernel<<<(ncopy + 255) / 256, 256, 0, stream>>>(y, dsty, ncopy);
  }
}
